// SlidingWindowAttention_25512105738575
// MI455X (gfx1250) — compile-verified, matrix operand order checked by reading, not run on hardware
//
#include <hip/hip_runtime.h>
#include <hip/hip_bf16.h>

// ---------------------------------------------------------------------------
// Types
// ---------------------------------------------------------------------------
typedef __bf16 bf16;
typedef __attribute__((ext_vector_type(8)))  __bf16 bf16x8;
typedef __attribute__((ext_vector_type(16))) __bf16 bf16x16;
typedef __attribute__((ext_vector_type(8)))  float  f32x8;
typedef __attribute__((ext_vector_type(4)))  unsigned u32x4;
typedef __attribute__((ext_vector_type(8)))  unsigned u32x8;

#define B_    2
#define S_    2048
#define HID_  2048
#define NH_   16
#define NKV_  4
#define HD_   128
#define WIN_  512

__device__ __forceinline__ f32x8 wmma_bf16(bf16x16 a, bf16x16 b, f32x8 c) {
    return __builtin_amdgcn_wmma_f32_16x16x32_bf16(false, a, false, b, (short)0, c,
                                                   false, false);
}

// ---------------------------------------------------------------------------
// Tensor Data Mover: 2D tile (global, row-major) -> LDS, with LDS row padding.
// D# per CDNA5 ISA ch.8 (group0 128b in 4 SGPRs, group1 256b in 8 SGPRs;
// groups 2/3 omitted => 2D).  Issued once per calling wave; TENSORcnt-tracked.
// lds offset = low 32 bits of the flat address (LDS aperture keeps the offset
// in addr[31:0]).  tile_w / row_stride in elements (data_size = 2 bytes).
// ---------------------------------------------------------------------------
__device__ __forceinline__ void tdm_load_2d(void* ldsPtr, const void* gPtr,
                                            unsigned tile_w, unsigned tile_h,
                                            unsigned row_stride, unsigned pad_ivl,
                                            unsigned pad_amt) {
    unsigned long long ga = (unsigned long long)(uintptr_t)gPtr;
    unsigned lds_off = (unsigned)(uintptr_t)ldsPtr;  // flat -> LDS byte offset
    u32x4 g0;
    g0[0] = 1u;                                         // count=1, user mode
    g0[1] = lds_off;                                    // lds_addr
    g0[2] = (unsigned)ga;                               // global_addr[31:0]
    g0[3] = (unsigned)((ga >> 32) & 0x01FFFFFFu)        // global_addr[56:32]
            | 0x80000000u;                              // type=2 ("image")
    u32x8 g1;
    g1[0] = (1u << 16)                                  // data_size = 2 bytes
            | (1u << 20)                                // pad_enable
            | (pad_ivl << 22) | (pad_amt << 25);        // LDS pad interval/amount
    g1[1] = (tile_w & 0xFFFFu) << 16;                   // tensor_dim0[15:0]
    g1[2] = (tile_h & 0xFFFFu) << 16;                   // tensor_dim1[15:0]
    g1[3] = (tile_w & 0xFFFFu) << 16;                   // tile_dim0
    g1[4] = (tile_h & 0xFFFFu);                         // tile_dim1 (tile_dim2=0)
    g1[5] = row_stride;                                 // tensor_dim0_stride[31:0]
    g1[6] = 0u;
    g1[7] = 0u;
    asm volatile("tensor_load_to_lds %0, %1" ::"s"(g0), "s"(g1) : "memory");
}

// A-style fragment (and B when the tile is stored [n][k]): lane holds row
// (lane&15); data = 8 contiguous elems at k0+8*hi and k0+16+8*hi (ISA 7.12.2).
__device__ __forceinline__ bf16x16 frag_row(const bf16* base, int ld, int row0, int k0,
                                            int lane) {
    int m = lane & 15, hi = lane >> 4;
    const bf16* p = base + (row0 + m) * ld + k0 + 8 * hi;
    bf16x8 a = *(const bf16x8*)p;
    bf16x8 c = *(const bf16x8*)(p + 16);
    return __builtin_shufflevector(a, c, 0, 1, 2, 3, 4, 5, 6, 7, 8, 9, 10, 11, 12, 13,
                                   14, 15);
}

// ---------------------------------------------------------------------------
// 1) fp32 -> bf16 conversion (grid-stride)
// ---------------------------------------------------------------------------
__global__ void k_cast_bf16(const float* __restrict__ src, bf16* __restrict__ dst,
                            int n) {
    for (int i = blockIdx.x * blockDim.x + threadIdx.x; i < n;
         i += gridDim.x * blockDim.x)
        dst[i] = (bf16)src[i];
}

// ---------------------------------------------------------------------------
// 2) Fused QKV projection + RoPE + relayout.
//    C = X @ Wqkv^T  (M=4096, N=3072, K=2048), block tile 128x128.
//    Each N-block is exactly one head (HD=128; region splits at 2048/2560 are
//    128-aligned), so the rotate-half pair (d, d+64) is block-local.  The fp32
//    tile is staged in LDS (union with the A/W tiles), RoPE'd, and written as
//    bf16 directly into the attention layouts:
//      Q -> [B,H,S,D], K -> [B,KV,S,D], V -> [B,KV,D,S] (transposed!).
// ---------------------------------------------------------------------------
struct GemmTiles {
    bf16 a[2][128 * 72];
    bf16 w[2][128 * 72];
};
union QkvSmem {
    GemmTiles t;          // 73,728 B
    float     c[128 * 132];  // 67,584 B
};

__global__ __launch_bounds__(256) void k_gemm_qkv(const bf16* __restrict__ X,
                                                  const bf16* __restrict__ W,
                                                  const float* __restrict__ cosT,
                                                  const float* __restrict__ sinT,
                                                  bf16* __restrict__ Qr,
                                                  bf16* __restrict__ Kr,
                                                  bf16* __restrict__ Vt) {
    __shared__ QkvSmem sm;
    const int K = HID_;
    const int tid = threadIdx.x, lane = tid & 31, wave = tid >> 5;
    const int wm = wave & 3, wn = wave >> 2;
    const int m0 = blockIdx.y * 128, n0 = blockIdx.x * 128;
    const int hi = lane >> 4, mlo = lane & 15;

    f32x8 acc[2][4];
#pragma unroll
    for (int i = 0; i < 2; ++i)
#pragma unroll
        for (int j = 0; j < 4; ++j) acc[i][j] = (f32x8){0, 0, 0, 0, 0, 0, 0, 0};

    const bf16* aT = X + (size_t)m0 * K;
    const bf16* wT = W + (size_t)n0 * K;

    if (wave == 0) {
        tdm_load_2d(&sm.t.a[0][0], aT, 64, 128, (unsigned)K, 4, 3);
        tdm_load_2d(&sm.t.w[0][0], wT, 64, 128, (unsigned)K, 4, 3);
        __builtin_amdgcn_s_wait_tensorcnt(0);
    }
    __syncthreads();

    int cur = 0;
    for (int kb = 0; kb < K; kb += 64) {
        if (wave == 0 && kb + 64 < K) {
            tdm_load_2d(&sm.t.a[cur ^ 1][0], aT + kb + 64, 64, 128, (unsigned)K, 4, 3);
            tdm_load_2d(&sm.t.w[cur ^ 1][0], wT + kb + 64, 64, 128, (unsigned)K, 4, 3);
        }
#pragma unroll
        for (int ks = 0; ks < 2; ++ks) {
            bf16x16 af[2], wf[4];
#pragma unroll
            for (int i = 0; i < 2; ++i)
                af[i] = frag_row(&sm.t.a[cur][0], 72, wm * 32 + i * 16, ks * 32, lane);
#pragma unroll
            for (int j = 0; j < 4; ++j)
                wf[j] = frag_row(&sm.t.w[cur][0], 72, wn * 64 + j * 16, ks * 32, lane);
#pragma unroll
            for (int i = 0; i < 2; ++i)
#pragma unroll
                for (int j = 0; j < 4; ++j)
                    acc[i][j] = wmma_bf16(af[i], wf[j], acc[i][j]);
        }
        if (wave == 0 && kb + 64 < K) __builtin_amdgcn_s_wait_tensorcnt(0);
        __syncthreads();
        cur ^= 1;
    }

    // ---- epilogue: stage fp32 tile in LDS (reusing the tile space) ----
#pragma unroll
    for (int i = 0; i < 2; ++i)
#pragma unroll
        for (int j = 0; j < 4; ++j) {
            const int col = wn * 64 + j * 16 + mlo;
#pragma unroll
            for (int r = 0; r < 8; ++r)
                sm.c[(wm * 32 + i * 16 + r + 8 * hi) * 132 + col] = acc[i][j][r];
        }
    __syncthreads();

    if (n0 < 2560) {  // Q or K region: RoPE, row-major bf16 store
        const bool isK = (n0 >= 2048);
        bf16* dst = isK ? (Kr + ((size_t)((n0 - 2048) >> 7)) * S_ * HD_)
                        : (Qr + ((size_t)(n0 >> 7)) * S_ * HD_);
        const size_t headStride = (size_t)(isK ? NKV_ : NH_) * S_ * HD_;
        for (int e = tid; e < 128 * 128; e += 256) {
            const int col = e & 127, row = e >> 7;
            const int tok = m0 + row, s = tok & (S_ - 1), b = tok >> 11;
            const float c   = sm.c[row * 132 + col];
            const float rot = (col < 64) ? -sm.c[row * 132 + col + 64]
                                         : sm.c[row * 132 + col - 64];
            const float res = c * cosT[s * HD_ + col] + rot * sinT[s * HD_ + col];
            dst[(size_t)b * headStride + (size_t)s * HD_ + col] = (bf16)res;
        }
    } else {  // V region: transposed store [B,KV,D,S], s thread-fastest (coalesced)
        const int kv = (n0 - 2560) >> 7;
        for (int e = tid; e < 128 * 128; e += 256) {
            const int row = e & 127;  // s-local, fastest across threads
            const int col = e >> 7;   // d
            const int tok = m0 + row, s = tok & (S_ - 1), b = tok >> 11;
            Vt[(((size_t)(b * NKV_ + kv)) * HD_ + col) * S_ + s] =
                (bf16)sm.c[row * 132 + col];
        }
    }
}

// ---------------------------------------------------------------------------
// 3) Generic bf16 WMMA GEMM (used for the output projection):
//    C[M,N] (f32) = A[M,K] @ W[N,K]^T, TDM double-buffered, K-step 64.
// ---------------------------------------------------------------------------
__global__ __launch_bounds__(256) void k_gemm_bf16(const bf16* __restrict__ A,
                                                   const bf16* __restrict__ W,
                                                   float* __restrict__ C, int M, int N,
                                                   int K) {
    __shared__ bf16 As[2][128 * 72];
    __shared__ bf16 Ws[2][128 * 72];

    const int tid = threadIdx.x, lane = tid & 31, wave = tid >> 5;
    const int wm = wave & 3, wn = wave >> 2;
    const int m0 = blockIdx.y * 128, n0 = blockIdx.x * 128;
    const int hi = lane >> 4, mlo = lane & 15;

    f32x8 acc[2][4];
#pragma unroll
    for (int i = 0; i < 2; ++i)
#pragma unroll
        for (int j = 0; j < 4; ++j) acc[i][j] = (f32x8){0, 0, 0, 0, 0, 0, 0, 0};

    const bf16* aT = A + (size_t)m0 * K;
    const bf16* wT = W + (size_t)n0 * K;

    if (wave == 0) {
        tdm_load_2d(&As[0][0], aT, 64, 128, (unsigned)K, 4, 3);
        tdm_load_2d(&Ws[0][0], wT, 64, 128, (unsigned)K, 4, 3);
        __builtin_amdgcn_s_wait_tensorcnt(0);
    }
    __syncthreads();

    int cur = 0;
    for (int kb = 0; kb < K; kb += 64) {
        if (wave == 0 && kb + 64 < K) {
            tdm_load_2d(&As[cur ^ 1][0], aT + kb + 64, 64, 128, (unsigned)K, 4, 3);
            tdm_load_2d(&Ws[cur ^ 1][0], wT + kb + 64, 64, 128, (unsigned)K, 4, 3);
        }
#pragma unroll
        for (int ks = 0; ks < 2; ++ks) {
            bf16x16 af[2], wf[4];
#pragma unroll
            for (int i = 0; i < 2; ++i)
                af[i] = frag_row(&As[cur][0], 72, wm * 32 + i * 16, ks * 32, lane);
#pragma unroll
            for (int j = 0; j < 4; ++j)
                wf[j] = frag_row(&Ws[cur][0], 72, wn * 64 + j * 16, ks * 32, lane);
#pragma unroll
            for (int i = 0; i < 2; ++i)
#pragma unroll
                for (int j = 0; j < 4; ++j)
                    acc[i][j] = wmma_bf16(af[i], wf[j], acc[i][j]);
        }
        if (wave == 0 && kb + 64 < K) __builtin_amdgcn_s_wait_tensorcnt(0);
        __syncthreads();
        cur ^= 1;
    }

#pragma unroll
    for (int i = 0; i < 2; ++i)
#pragma unroll
        for (int j = 0; j < 4; ++j) {
            const int gcol = n0 + wn * 64 + j * 16 + mlo;
#pragma unroll
            for (int r = 0; r < 8; ++r) {
                const int grow = m0 + wm * 32 + i * 16 + r + 8 * hi;
                C[(size_t)grow * N + gcol] = acc[i][j][r];
            }
        }
}

// ---------------------------------------------------------------------------
// 4) Flash attention, sliding window 512.
//    Block: 128 queries x 1 head; 8 waves each own 16 query rows.
//    K tiles [64 keys x 128 d] and V tiles [128 d x 64 keys] (pre-transposed)
//    are TDM double-buffered; one barrier per key tile.
// ---------------------------------------------------------------------------
__global__ __launch_bounds__(256) void k_attn(const bf16* __restrict__ Qr,
                                              const bf16* __restrict__ Kr,
                                              const bf16* __restrict__ Vt,
                                              bf16* __restrict__ Obf) {
    __shared__ bf16 Ks[2][64 * 136];   // rows 256B -> TDM pad ivl=5, amt=3
    __shared__ bf16 Vs[2][128 * 72];   // rows 128B -> TDM pad ivl=4, amt=3
    __shared__ bf16 Ps[8][16 * 72];

    const int qt = blockIdx.x, h = blockIdx.y, b = blockIdx.z;
    const int q0 = qt * 128;
    const int kvh = h >> 2;  // N_REP = 4
    const int tid = threadIdx.x, lane = tid & 31, wave = tid >> 5;
    const int mlo = lane & 15, hi = lane >> 4;
    const float scale = 0.08838834764831845f;  // 1/sqrt(128)

    const bf16* Qbase = Qr + ((size_t)(b * NH_ + h)) * S_ * HD_;
    const bf16* Kbase = Kr + ((size_t)(b * NKV_ + kvh)) * S_ * HD_;
    const bf16* Vbase = Vt + ((size_t)(b * NKV_ + kvh)) * HD_ * S_;  // [d][s]

    // Preload Q fragments (A layout: two contiguous 16B runs per lane).
    bf16x16 qf[4];
    {
        const int qrow = q0 + wave * 16 + mlo;
#pragma unroll
        for (int kk = 0; kk < 4; ++kk) {
            const bf16* p = Qbase + (size_t)qrow * HD_ + kk * 32 + 8 * hi;
            bf16x8 a = *(const bf16x8*)p;
            bf16x8 c = *(const bf16x8*)(p + 16);
            qf[kk] = __builtin_shufflevector(a, c, 0, 1, 2, 3, 4, 5, 6, 7, 8, 9, 10, 11,
                                             12, 13, 14, 15);
        }
    }

    f32x8 oacc[8];
    float mrow[8], lrow[8];
#pragma unroll
    for (int n = 0; n < 8; ++n) {
        oacc[n] = (f32x8){0, 0, 0, 0, 0, 0, 0, 0};
        mrow[n] = -3.0e38f;
        lrow[n] = 0.0f;
    }

    const int kt_lo = (q0 > 511) ? ((q0 - 511) >> 6) : 0;
    const int kt_hi = (q0 + 127) >> 6;

    if (wave == 0) {
        const int k0 = kt_lo << 6;
        tdm_load_2d(&Ks[0][0], Kbase + (size_t)k0 * HD_, 128, 64, 128, 5, 3);
        tdm_load_2d(&Vs[0][0], Vbase + k0, 64, 128, (unsigned)S_, 4, 3);
        __builtin_amdgcn_s_wait_tensorcnt(0);
    }
    __syncthreads();

    int cur = 0;
    for (int kt = kt_lo; kt <= kt_hi; ++kt) {
        const int k0 = kt << 6;
        if (wave == 0 && kt < kt_hi) {  // async DMA of next K/V tiles
            tdm_load_2d(&Ks[cur ^ 1][0], Kbase + (size_t)(k0 + 64) * HD_, 128, 64, 128,
                        5, 3);
            tdm_load_2d(&Vs[cur ^ 1][0], Vbase + k0 + 64, 64, 128, (unsigned)S_, 4, 3);
        }

        // scores: S = Q K^T  (4 column subtiles x 4 K-steps)
        f32x8 sacc[4];
#pragma unroll
        for (int n = 0; n < 4; ++n) {
            f32x8 a = (f32x8){0, 0, 0, 0, 0, 0, 0, 0};
#pragma unroll
            for (int kk = 0; kk < 4; ++kk)
                a = wmma_bf16(qf[kk], frag_row(&Ks[cur][0], 136, n * 16, kk * 32, lane),
                              a);
            sacc[n] = a;
        }

        // mask + online softmax (row = r + 8*hi, col = lane&15)
        const int ibase = q0 + wave * 16 + 8 * hi;
#pragma unroll
        for (int r = 0; r < 8; ++r) {
            const int i = ibase + r;
            float rm = -3.0e38f;
#pragma unroll
            for (int n = 0; n < 4; ++n) {
                const int j = k0 + n * 16 + mlo;
                float s = sacc[n][r] * scale;
                s = ((j <= i) && (j > i - WIN_)) ? s : -3.0e38f;
                sacc[n][r] = s;
                rm = fmaxf(rm, s);
            }
            rm = fmaxf(rm, __shfl_xor(rm, 1, 32));
            rm = fmaxf(rm, __shfl_xor(rm, 2, 32));
            rm = fmaxf(rm, __shfl_xor(rm, 4, 32));
            rm = fmaxf(rm, __shfl_xor(rm, 8, 32));
            const float newm = fmaxf(mrow[r], rm);
            const float corr = __expf(mrow[r] - newm);
            mrow[r] = newm;
            float rs = 0.0f;
#pragma unroll
            for (int n = 0; n < 4; ++n) {
                const float p = __expf(sacc[n][r] - newm);
                sacc[n][r] = p;
                rs += p;
            }
            rs += __shfl_xor(rs, 1, 32);
            rs += __shfl_xor(rs, 2, 32);
            rs += __shfl_xor(rs, 4, 32);
            rs += __shfl_xor(rs, 8, 32);
            lrow[r] = lrow[r] * corr + rs;
#pragma unroll
            for (int nn = 0; nn < 8; ++nn) oacc[nn][r] *= corr;
#pragma unroll
            for (int n = 0; n < 4; ++n)
                Ps[wave][(r + 8 * hi) * 72 + n * 16 + mlo] = (bf16)sacc[n][r];
        }
        // No barrier: each wave reads back only Ps[wave]; per-wave LDS ops are
        // in-order (DScnt), so the P-store -> frag-load RAW is wave-local.

        // O += P @ V   (K = 64 -> 2 steps, 8 output column subtiles)
        // V tile is [d][key] so the B fragment is a contiguous frag_row.
#pragma unroll
        for (int kk = 0; kk < 2; ++kk) {
            bf16x16 pf = frag_row(&Ps[wave][0], 72, 0, kk * 32, lane);
#pragma unroll
            for (int nn = 0; nn < 8; ++nn)
                oacc[nn] = wmma_bf16(
                    pf, frag_row(&Vs[cur][0], 72, nn * 16, kk * 32, lane), oacc[nn]);
        }

        if (wave == 0 && kt < kt_hi) __builtin_amdgcn_s_wait_tensorcnt(0);
        __syncthreads();  // everyone done with cur; next tiles published
        cur ^= 1;
    }

    // epilogue: O / l  -> Obf[B*S, H*D]
#pragma unroll
    for (int r = 0; r < 8; ++r) {
        const int srow = q0 + wave * 16 + r + 8 * hi;
        const float inv = 1.0f / lrow[r];
        const size_t off = ((size_t)(b * S_ + srow)) * (NH_ * HD_) + h * HD_;
#pragma unroll
        for (int nn = 0; nn < 8; ++nn)
            Obf[off + nn * 16 + mlo] = (bf16)(oacc[nn][r] * inv);
    }
}

// ---------------------------------------------------------------------------
// launch
// ---------------------------------------------------------------------------
extern "C" void kernel_launch(void* const* d_in, const int* in_sizes, int n_in,
                              void* d_out, int out_size, void* d_ws, size_t ws_size,
                              hipStream_t stream) {
    const float* x    = (const float*)d_in[0];
    const float* cosT = (const float*)d_in[1];
    const float* sinT = (const float*)d_in[2];
    const float* wq   = (const float*)d_in[3];
    const float* wk   = (const float*)d_in[4];
    const float* wv   = (const float*)d_in[5];
    const float* wo   = (const float*)d_in[6];

    char* ws = (char*)d_ws;
    // workspace layout (bytes)
    bf16* Xbf  = (bf16*)(ws + 0);                       // 4096x2048       16 MB
    bf16* Wqkv = (bf16*)(ws + 16777216);                // 3072x2048       12 MB
    bf16* Wobf = (bf16*)(ws + 29360128);                // 2048x2048        8 MB
    bf16* Qr   = (bf16*)(ws + 37748736);                // [B,H,S,D]       16 MB
    bf16* Kr   = (bf16*)(ws + 54525952);                // [B,KV,S,D]       4 MB
    bf16* Vt   = (bf16*)(ws + 58720256);                // [B,KV,D,S]       4 MB
    bf16* Obf  = (bf16*)(ws + 62914560);                // 4096x2048       16 MB

    const int NX  = B_ * S_ * HID_;    // 8388608
    const int NWQ = NH_ * HD_ * HID_;  // 4194304
    const int NWK = NKV_ * HD_ * HID_; // 1048576

    // 1) casts (wq|wk|wv packed so QKV is one GEMM)
    k_cast_bf16<<<4096, 256, 0, stream>>>(x, Xbf, NX);
    k_cast_bf16<<<4096, 256, 0, stream>>>(wq, Wqkv, NWQ);
    k_cast_bf16<<<2048, 256, 0, stream>>>(wk, Wqkv + NWQ, NWK);
    k_cast_bf16<<<2048, 256, 0, stream>>>(wv, Wqkv + NWQ + NWK, NWK);
    k_cast_bf16<<<4096, 256, 0, stream>>>(wo, Wobf, NWQ);

    // 2) fused QKV projection + RoPE + relayout (V stored transposed)
    k_gemm_qkv<<<dim3(3072 / 128, 4096 / 128), 256, 0, stream>>>(Xbf, Wqkv, cosT, sinT,
                                                                 Qr, Kr, Vt);

    // 3) sliding-window flash attention
    k_attn<<<dim3(S_ / 128, NH_, B_), 256, 0, stream>>>(Qr, Kr, Vt, Obf);

    // 4) output projection -> fp32 d_out
    k_gemm_bf16<<<dim3(2048 / 128, 4096 / 128), 256, 0, stream>>>(
        Obf, Wobf, (float*)d_out, 4096, 2048, 2048);
}
